// BiLSTM_CRF_15083925143996
// MI455X (gfx1250) — compile-verified, matrix operand order checked by reading, not run on hardware
//
#include <hip/hip_runtime.h>
#include <hip/hip_bf16.h>

// ---------------------------------------------------------------------------
// BiLSTM-CRF for MI455X (gfx1250, wave32).
//   L=2048 tokens, E=512, HD=512 per direction, H=1024, T=48 tags.
// Input projections -> v_wmma_f32_16x16x32_bf16 tiled GEMM, double-buffered
//   LDS staged via GLOBAL_LOAD_ASYNC_TO_LDS (ASYNCcnt) when available.
// Recurrences -> latency-bound VALU kernels, h broadcast via LDS, c in regs.
// Viterbi -> single-workgroup max-plus scan + backtrack.
// ---------------------------------------------------------------------------

typedef __attribute__((ext_vector_type(16))) __bf16 v16bf;
typedef __attribute__((ext_vector_type(8)))  float  v8f;
typedef int v4i __attribute__((vector_size(16)));   // matches builtin param type

#define LSEQ 2048
#define EDIM 512
#define HDIM 512       // per-direction hidden
#define HTOT 1024
#define GDIM 2048      // 4*HDIM
#define TAGS 48
#define TSTART 46
#define TSTOP  47
#define NEGV  (-10000.0f)

// async-to-LDS path (gfx1250): guard on builtin availability so the kernel
// still compiles (sync fallback) on toolchains without these builtins.
#if defined(__has_builtin)
#if __has_builtin(__builtin_amdgcn_global_load_async_to_lds_b128) && \
    __has_builtin(__builtin_amdgcn_s_wait_asynccnt)
#define USE_ASYNC_LDS 1
#endif
#endif
#ifndef USE_ASYNC_LDS
#define USE_ASYNC_LDS 0
#endif

#if USE_ASYNC_LDS
#define GLOBAL_AS __attribute__((address_space(1)))
#define LDS_AS    __attribute__((address_space(3)))
typedef GLOBAL_AS v4i* gv4i_ptr;
typedef LDS_AS    v4i* lv4i_ptr;
#endif

// -------------------------------- helpers ----------------------------------

__device__ __forceinline__ unsigned short f2bf(float f) {
  unsigned int u = __float_as_uint(f);
  unsigned int r = (u + 0x7FFFu + ((u >> 16) & 1u)) >> 16;
  return (unsigned short)r;
}

__device__ __forceinline__ float sigf(float x) {
  return 1.0f / (1.0f + __expf(-x));
}

// ----------------------------- small kernels -------------------------------

__global__ void k_cast_bf16(const float* __restrict__ src,
                            unsigned short* __restrict__ dst, int n) {
  int i = blockIdx.x * blockDim.x + threadIdx.x;
  if (i < n) dst[i] = f2bf(src[i]);
}

__global__ void k_bias_fuse(const float* __restrict__ b_ih,
                            const float* __restrict__ b_hh,
                            float* __restrict__ b, int n) {
  int i = blockIdx.x * blockDim.x + threadIdx.x;
  if (i < n) b[i] = b_ih[i] + b_hh[i];
}

// Embedding gather + bf16 downconvert: one block per token.
__global__ __launch_bounds__(128)
void k_embed(const int* __restrict__ sent, const float* __restrict__ emb,
             unsigned short* __restrict__ out) {
  int t = blockIdx.x;
  const float* src = emb + (size_t)sent[t] * EDIM;
  unsigned short* dst = out + (size_t)t * EDIM;
#pragma unroll
  for (int i = 0; i < 4; ++i) {
    int e = threadIdx.x + i * 128;
    dst[e] = f2bf(src[e]);
  }
}

// ------------------------------ WMMA GEMM ----------------------------------
// C[M,N] = A[M,K](bf16) * B[N,K](bf16)^T + bias[N], fp32 out.
// Block: 256 threads = 8 waves; block tile 128(M) x 64(N); K step 32.
// Each wave owns a 32x32 output tile = 2x2 WMMA 16x16 accumulators.
// Double-buffered LDS; tiles staged with async-to-LDS DMA when available.

#define TM 128
#define TN 64
#define TK 32
#define LDK 40  // padded LDS row stride (halves), 16B aligned, conflict-reducing

__global__ __launch_bounds__(256)
void k_gemm_bf16(const unsigned short* __restrict__ A,   // [M,K]
                 const unsigned short* __restrict__ B,   // [N,K]
                 const float* __restrict__ bias,         // [N]
                 float* __restrict__ C,                   // [M,N]
                 int M, int N, int K) {
  __shared__ __align__(16) unsigned short As[2][TM][LDK];  // 2 x 10 KB
  __shared__ __align__(16) unsigned short Bs[2][TN][LDK];  // 2 x  5 KB

  const int tid  = threadIdx.x;
  const int lane = tid & 31;
  const int wave = tid >> 5;            // 0..7
  const int wm   = (wave >> 1) * 32;    // 0,32,64,96
  const int wn   = (wave & 1) * 32;     // 0,32
  const int mBase = blockIdx.x * TM;
  const int nBase = blockIdx.y * TN;

  v8f acc[2][2] = {};

  // cooperative-copy mapping
  const int a_row = tid >> 1;           // 0..127
  const int a_off = (tid & 1) * 16;     // halves
  const int b_row = tid >> 2;           // 0..63
  const int b_off = (tid & 3) * 8;      // halves

  const unsigned short* gA0 = A + (size_t)(mBase + a_row) * K + a_off;
  const unsigned short* gB0 = B + (size_t)(nBase + b_row) * K + b_off;

  const int fm = lane & 15;             // M (or N) index inside a 16-tile
  const int kb = (lane >> 4) * 8;       // K sub-base per ISA A/B layouts

  const int nk = K / TK;

  // ---- stage copy: issue tile (kt) into buffer buf ----
  auto issue = [&](int buf, int kt) {
#if USE_ASYNC_LDS
    // per-lane 16B async DMA straight into LDS; tracked by ASYNCcnt
    __builtin_amdgcn_global_load_async_to_lds_b128(
        (gv4i_ptr)(gA0 + kt), (lv4i_ptr)(&As[buf][a_row][a_off]), 0, 0);
    __builtin_amdgcn_global_load_async_to_lds_b128(
        (gv4i_ptr)(gA0 + kt), (lv4i_ptr)(&As[buf][a_row][a_off]), 16, 0);
    __builtin_amdgcn_global_load_async_to_lds_b128(
        (gv4i_ptr)(gB0 + kt), (lv4i_ptr)(&Bs[buf][b_row][b_off]), 0, 0);
#else
    const uint4* gA = (const uint4*)(gA0 + kt);
    uint4* sA = (uint4*)(&As[buf][a_row][a_off]);
    sA[0] = gA[0];
    sA[1] = gA[1];
    *(uint4*)(&Bs[buf][b_row][b_off]) = *(const uint4*)(gB0 + kt);
    if (kt + TK < K) {
      __builtin_prefetch(gA0 + kt + TK, 0, 1);
      __builtin_prefetch(gB0 + kt + TK, 0, 1);
    }
#endif
  };

  issue(0, 0);

  for (int s = 0; s < nk; ++s) {
#if USE_ASYNC_LDS
    __builtin_amdgcn_s_wait_asynccnt(0);   // this wave's stage-s DMA done
#endif
    __syncthreads();                       // all waves' tiles visible
    if (s + 1 < nk) issue((s + 1) & 1, (s + 1) * TK);

    const int buf = s & 1;
    // fragments: A 16x32 (lane<16: M=lane,K={0..7,16..23}; lane>=16: K={8..15,24..31})
    v16bf afrag[2], bfrag[2];
#pragma unroll
    for (int i = 0; i < 2; ++i) {
      union { v16bf v; uint4 q[2]; } u;
      u.q[0] = *(const uint4*)(&As[buf][wm + i * 16 + fm][kb]);
      u.q[1] = *(const uint4*)(&As[buf][wm + i * 16 + fm][kb + 16]);
      afrag[i] = u.v;
    }
#pragma unroll
    for (int j = 0; j < 2; ++j) {
      union { v16bf v; uint4 q[2]; } u;
      u.q[0] = *(const uint4*)(&Bs[buf][wn + j * 16 + fm][kb]);
      u.q[1] = *(const uint4*)(&Bs[buf][wn + j * 16 + fm][kb + 16]);
      bfrag[j] = u.v;
    }
#pragma unroll
    for (int i = 0; i < 2; ++i)
#pragma unroll
      for (int j = 0; j < 2; ++j)
        acc[i][j] = __builtin_amdgcn_wmma_f32_16x16x32_bf16(
            false, afrag[i], false, bfrag[j], (short)0, acc[i][j], false, false);
  }

  // store: VGPR r -> M = r + (lane>>4)*8, N = lane&15 (per ISA C/D layout)
  const int cm = (lane >> 4) * 8;
  const int cn = lane & 15;
#pragma unroll
  for (int i = 0; i < 2; ++i)
#pragma unroll
    for (int j = 0; j < 2; ++j) {
      int nn = nBase + wn + j * 16 + cn;
      float bv = bias[nn];
#pragma unroll
      for (int r = 0; r < 8; ++r) {
        int mm = mBase + wm + i * 16 + cm + r;
        C[(size_t)mm * N + nn] = acc[i][j][r] + bv;
      }
    }
}

// --------------------------- LSTM recurrence -------------------------------
// One block (512 threads) per direction; grid.x = 2 runs fwd+rev concurrently.
// Thread j owns gate rows j (i), 512+j (f), 1024+j (g), 1536+j (o) and cell j.
// h broadcast via LDS each step; w_hh rows streamed from L2 (4MB resident).

__global__ __launch_bounds__(512)
void k_lstm2(const float* __restrict__ preF, const float* __restrict__ preR,
             const float* __restrict__ whhF, const float* __restrict__ whhR,
             const float* __restrict__ h0, const float* __restrict__ c0,
             int rowF, int rowR,
             float* __restrict__ out, int colF, int colR) {
  __shared__ __align__(16) float hbuf[HDIM];

  const bool rev = (blockIdx.x == 1);
  const float* pre = rev ? preR : preF;
  const float* whh = rev ? whhR : whhF;
  const int h0row  = rev ? rowR : rowF;
  const int col    = rev ? colR : colF;
  const int j = threadIdx.x;

  float cc = c0[h0row * HDIM + j];
  hbuf[j]  = h0[h0row * HDIM + j];
  __syncthreads();

  const float4* wi = (const float4*)(whh + (size_t)(j)        * HDIM);
  const float4* wf = (const float4*)(whh + (size_t)(512 + j)  * HDIM);
  const float4* wg = (const float4*)(whh + (size_t)(1024 + j) * HDIM);
  const float4* wo = (const float4*)(whh + (size_t)(1536 + j) * HDIM);
  const float4* h4 = (const float4*)hbuf;

  for (int s = 0; s < LSEQ; ++s) {
    const int t = rev ? (LSEQ - 1 - s) : s;
    const float* pr = pre + (size_t)t * GDIM;
    float ai = pr[j];
    float af = pr[512 + j];
    float ag = pr[1024 + j];
    float ao = pr[1536 + j];
#pragma unroll 4
    for (int k = 0; k < HDIM / 4; ++k) {
      float4 hv = h4[k];
      float4 a = wi[k], b = wf[k], g = wg[k], o = wo[k];
      ai += hv.x * a.x + hv.y * a.y + hv.z * a.z + hv.w * a.w;
      af += hv.x * b.x + hv.y * b.y + hv.z * b.z + hv.w * b.w;
      ag += hv.x * g.x + hv.y * g.y + hv.z * g.z + hv.w * g.w;
      ao += hv.x * o.x + hv.y * o.y + hv.z * o.z + hv.w * o.w;
    }
    float ig = sigf(ai);
    float fg = sigf(af);
    float gg = tanhf(ag);
    float og = sigf(ao);
    cc = fg * cc + ig * gg;
    float hn = og * tanhf(cc);
    out[(size_t)t * HTOT + col + j] = hn;
    __syncthreads();      // everyone done reading old h
    hbuf[j] = hn;
    __syncthreads();      // new h visible
  }
}

// ---------------------------- tag projection -------------------------------

__global__ __launch_bounds__(64)
void k_tagproj(const float* __restrict__ l1, const float* __restrict__ Wt,
               const float* __restrict__ bt, float* __restrict__ feats) {
  __shared__ __align__(16) float row[HTOT];
  const int t = blockIdx.x;
  for (int i = threadIdx.x; i < HTOT; i += 64) row[i] = l1[(size_t)t * HTOT + i];
  __syncthreads();
  if (threadIdx.x < TAGS) {
    const float4* w = (const float4*)(Wt + (size_t)threadIdx.x * HTOT);
    const float4* r4 = (const float4*)row;
    float acc = bt[threadIdx.x];
#pragma unroll 4
    for (int k = 0; k < HTOT / 4; ++k) {
      float4 a = r4[k], b = w[k];
      acc += a.x * b.x + a.y * b.y + a.z * b.z + a.w * b.w;
    }
    feats[t * TAGS + threadIdx.x] = acc;
  }
}

// -------------------------------- Viterbi ----------------------------------

__global__ __launch_bounds__(64)
void k_viterbi(const float* __restrict__ feats, const float* __restrict__ trans,
               int* __restrict__ bps, float* __restrict__ out) {
  __shared__ float fv[TAGS];
  __shared__ float tr[TAGS * TAGS];
  const int j = threadIdx.x;
  for (int i = j; i < TAGS * TAGS; i += 64) tr[i] = trans[i];
  if (j < TAGS) fv[j] = (j == TSTART) ? 0.0f : NEGV;
  __syncthreads();

  for (int t = 0; t < LSEQ; ++t) {
    float m = -3.4e38f;
    int bp = 0;
    if (j < TAGS) {
      const float* tj = &tr[j * TAGS];
#pragma unroll 8
      for (int p = 0; p < TAGS; ++p) {
        float s = fv[p] + tj[p];
        if (s > m) { m = s; bp = p; }     // first-max tie-break like argmax
      }
    }
    __syncthreads();
    if (j < TAGS) {
      fv[j] = m + feats[t * TAGS + j];
      bps[t * TAGS + j] = bp;
    }
    __syncthreads();
  }

  if (j == 0) {
    float best = -3.4e38f;
    int bt = 0;
    for (int p = 0; p < TAGS; ++p) {
      float s = fv[p] + tr[TSTOP * TAGS + p];
      if (s > best) { best = s; bt = p; }
    }
    out[0] = best;                 // score
    out[LSEQ] = (float)bt;         // path[L-1]
    int tag = bt;
    for (int t = LSEQ - 1; t >= 1; --t) {
      tag = bps[t * TAGS + tag];
      out[t] = (float)tag;         // path[t-1] at out[1 + (t-1)]
    }
  }
}

// ------------------------------- launcher ----------------------------------

extern "C" void kernel_launch(void* const* d_in, const int* in_sizes, int n_in,
                              void* d_out, int out_size, void* d_ws, size_t ws_size,
                              hipStream_t stream) {
  (void)in_sizes; (void)n_in; (void)out_size; (void)ws_size;

  const int*   sent   = (const int*)d_in[0];
  const float* emb    = (const float*)d_in[1];
  const float* wih0f  = (const float*)d_in[2];
  const float* whh0f  = (const float*)d_in[3];
  const float* bih0f  = (const float*)d_in[4];
  const float* bhh0f  = (const float*)d_in[5];
  const float* wih0r  = (const float*)d_in[6];
  const float* whh0r  = (const float*)d_in[7];
  const float* bih0r  = (const float*)d_in[8];
  const float* bhh0r  = (const float*)d_in[9];
  const float* wih1f  = (const float*)d_in[10];
  const float* whh1f  = (const float*)d_in[11];
  const float* bih1f  = (const float*)d_in[12];
  const float* bhh1f  = (const float*)d_in[13];
  const float* wih1r  = (const float*)d_in[14];
  const float* whh1r  = (const float*)d_in[15];
  const float* bih1r  = (const float*)d_in[16];
  const float* bhh1r  = (const float*)d_in[17];
  const float* h0     = (const float*)d_in[18];
  const float* c0     = (const float*)d_in[19];
  const float* Wtag   = (const float*)d_in[20];
  const float* btag   = (const float*)d_in[21];
  const float* transi = (const float*)d_in[22];
  float* outp = (float*)d_out;

  // ---- workspace carve-up (deterministic) ----
  char* base = (char*)d_ws;
  size_t o = 0;
  auto carve = [&](size_t bytes) -> char* {
    char* p = base + o;
    o = (o + bytes + 255) & ~(size_t)255;
    return p;
  };
  unsigned short* x0b    = (unsigned short*)carve((size_t)LSEQ * EDIM * 2);
  unsigned short* w0f_b  = (unsigned short*)carve((size_t)GDIM * EDIM * 2);
  unsigned short* w0r_b  = (unsigned short*)carve((size_t)GDIM * EDIM * 2);
  unsigned short* w1f_b  = (unsigned short*)carve((size_t)GDIM * HTOT * 2);
  unsigned short* w1r_b  = (unsigned short*)carve((size_t)GDIM * HTOT * 2);
  float* bias0f = (float*)carve(GDIM * 4);
  float* bias0r = (float*)carve(GDIM * 4);
  float* bias1f = (float*)carve(GDIM * 4);
  float* bias1r = (float*)carve(GDIM * 4);
  float* preA   = (float*)carve((size_t)LSEQ * GDIM * 4);   // fwd pre (reused L0/L1)
  float* preB   = (float*)carve((size_t)LSEQ * GDIM * 4);   // rev pre (reused L0/L1)
  float* l0     = (float*)carve((size_t)LSEQ * HTOT * 4);
  unsigned short* l0b = (unsigned short*)carve((size_t)LSEQ * HTOT * 2);
  float* l1     = (float*)carve((size_t)LSEQ * HTOT * 4);
  float* feats  = (float*)carve((size_t)LSEQ * TAGS * 4);
  int*   bps    = (int*)carve((size_t)LSEQ * TAGS * 4);

  // ---- 1. cast w_ih weights to bf16 ----
  {
    int n0 = GDIM * EDIM, n1 = GDIM * HTOT;
    k_cast_bf16<<<(n0 + 255) / 256, 256, 0, stream>>>(wih0f, w0f_b, n0);
    k_cast_bf16<<<(n0 + 255) / 256, 256, 0, stream>>>(wih0r, w0r_b, n0);
    k_cast_bf16<<<(n1 + 255) / 256, 256, 0, stream>>>(wih1f, w1f_b, n1);
    k_cast_bf16<<<(n1 + 255) / 256, 256, 0, stream>>>(wih1r, w1r_b, n1);
  }
  // ---- 2. fuse biases ----
  k_bias_fuse<<<GDIM / 256, 256, 0, stream>>>(bih0f, bhh0f, bias0f, GDIM);
  k_bias_fuse<<<GDIM / 256, 256, 0, stream>>>(bih0r, bhh0r, bias0r, GDIM);
  k_bias_fuse<<<GDIM / 256, 256, 0, stream>>>(bih1f, bhh1f, bias1f, GDIM);
  k_bias_fuse<<<GDIM / 256, 256, 0, stream>>>(bih1r, bhh1r, bias1r, GDIM);

  // ---- 3. embedding gather (bf16) ----
  k_embed<<<LSEQ, 128, 0, stream>>>(sent, emb, x0b);

  // ---- 4. layer-0 input projections (WMMA) ----
  dim3 g0(LSEQ / TM, GDIM / TN);
  k_gemm_bf16<<<g0, 256, 0, stream>>>(x0b, w0f_b, bias0f, preA, LSEQ, GDIM, EDIM);
  k_gemm_bf16<<<g0, 256, 0, stream>>>(x0b, w0r_b, bias0r, preB, LSEQ, GDIM, EDIM);

  // ---- 5. layer-0 recurrence (fwd + rev concurrently) ----
  k_lstm2<<<2, 512, 0, stream>>>(preA, preB, whh0f, whh0r, h0, c0, 0, 1, l0, 0, HDIM);

  // ---- 6. l0 -> bf16 ----
  k_cast_bf16<<<(LSEQ * HTOT + 255) / 256, 256, 0, stream>>>(l0, l0b, LSEQ * HTOT);

  // ---- 7. layer-1 input projections (WMMA) ----
  k_gemm_bf16<<<g0, 256, 0, stream>>>(l0b, w1f_b, bias1f, preA, LSEQ, GDIM, HTOT);
  k_gemm_bf16<<<g0, 256, 0, stream>>>(l0b, w1r_b, bias1r, preB, LSEQ, GDIM, HTOT);

  // ---- 8. layer-1 recurrence ----
  k_lstm2<<<2, 512, 0, stream>>>(preA, preB, whh1f, whh1r, h0, c0, 2, 3, l1, 0, HDIM);

  // ---- 9. tag projection ----
  k_tagproj<<<LSEQ, 64, 0, stream>>>(l1, Wtag, btag, feats);

  // ---- 10. Viterbi + backtrack -> d_out ----
  k_viterbi<<<1, 64, 0, stream>>>(feats, transi, bps, outp);
}
